// LinearAttention_38835094290658
// MI455X (gfx1250) — matrix operand order repaired and checked by reading, not compiled, not run on hardware
//
#include <hip/hip_runtime.h>
#include <hip/hip_bf16.h>
#include <math.h>

// ---------------------------------------------------------------------------
// Linear attention (FAVOR-style) on MI455X / gfx1250, bf16 WMMA pipeline.
// B=8, L=4096, D=1024, E=256.
// Stages: prep (f32->bf16 + transposes), proj+exp, kv = x^T k', out = q' kv * z
// All GEMM stages use v_wmma_f32_16x16x32_bf16, ping-pong double buffered
// (manual 2x unroll -> no register-rotation copies).
// ---------------------------------------------------------------------------

typedef __bf16 bf16;
typedef __attribute__((ext_vector_type(16))) __bf16 v16bf;
typedef __attribute__((ext_vector_type(8)))  __bf16 v8bf;
typedef __attribute__((ext_vector_type(8)))  float  v8f;

static constexpr int BB = 8;
static constexpr int LL = 4096;
static constexpr int DD = 1024;
static constexpr int EE = 256;

// ---- WMMA helper -----------------------------------------------------------
__device__ __forceinline__ v8f wmma_bf16(v16bf a, v16bf b, v8f c) {
  return __builtin_amdgcn_wmma_f32_16x16x32_bf16(
      /*neg_a=*/false, a, /*neg_b=*/false, b,
      /*c_mod=*/(short)0, c, /*reuse_a=*/false, /*reuse_b=*/false);
}

// A-fragment (16x32 bf16) from a row-major [M x K] source.
// Lane (m = lane&15, half = lane>>4) holds K = 16*half + j, contiguous 32B.
__device__ __forceinline__ v16bf load_afrag(const bf16* p /* row base + k0 + 16*half */) {
  return *(const v16bf*)(p);
}

// B-fragment (32x16 bf16) from B^T stored row-major [N x K]:
// lane (n = lane&15, half = lane>>4) holds K = 16*half + j, contiguous 32B.
__device__ __forceinline__ v16bf load_bfrag(const bf16* p /* BT row + k0 + 16*half */) {
  return *(const v16bf*)(p);
}

// ---- prep: x f32 -> bf16 in [B,L,D] and transposed [B,D,L] -----------------
__global__ __launch_bounds__(256) void prep_x_kernel(const float* __restrict__ x,
                                                     bf16* __restrict__ xbf,
                                                     bf16* __restrict__ xT) {
  __shared__ bf16 t[32][33];
  const int b  = blockIdx.z;
  const int gd = blockIdx.x * 32;
  const int gl = blockIdx.y * 32;
  const int tx = threadIdx.x, ty = threadIdx.y;  // (32, 8)
#pragma unroll
  for (int i = 0; i < 4; ++i) {
    const int l = gl + ty + i * 8, d = gd + tx;
    // x is consumed exactly once -> non-temporal, keep L2 for bf16 tensors
    const float v = __builtin_nontemporal_load(&x[((size_t)b * LL + l) * DD + d]);
    const bf16 h = (bf16)v;
    xbf[((size_t)b * LL + l) * DD + d] = h;
    t[ty + i * 8][tx] = h;
  }
  __syncthreads();
#pragma unroll
  for (int i = 0; i < 4; ++i) {
    const int d = gd + ty + i * 8, l = gl + tx;
    xT[((size_t)b * DD + d) * LL + l] = t[tx][ty + i * 8];
  }
}

// ---- prep: proj [D,E] f32 -> projT [E,D] bf16 ------------------------------
__global__ __launch_bounds__(256) void prep_proj_kernel(const float* __restrict__ qp,
                                                        const float* __restrict__ kp,
                                                        bf16* __restrict__ qT,
                                                        bf16* __restrict__ kT) {
  __shared__ bf16 t[32][33];
  const float* src = blockIdx.z ? kp : qp;
  bf16* dst = blockIdx.z ? kT : qT;
  const int ge = blockIdx.x * 32;  // e
  const int gd = blockIdx.y * 32;  // d
  const int tx = threadIdx.x, ty = threadIdx.y;
#pragma unroll
  for (int i = 0; i < 4; ++i) {
    const int d = gd + ty + i * 8;
    t[ty + i * 8][tx] = (bf16)src[(size_t)d * EE + ge + tx];
  }
  __syncthreads();
#pragma unroll
  for (int i = 0; i < 4; ++i) {
    const int e = ge + ty + i * 8;
    dst[(size_t)e * DD + gd + tx] = t[tx][ty + i * 8];
  }
}

// ---- stage 1: p' = exp(x @ proj); q' row-major + rowsum; k' transposed -----
// block = 128 threads (4 waves); block tile = 32 rows (L) x 256 cols (E).
// Each wave: 2 A-row-fragments x 4 B-tiles = 8 WMMA per K-step, ping-pong
// double buffered over K (no rotation copies).
__global__ __launch_bounds__(128) void proj_exp_kernel(
    const bf16* __restrict__ xbf,     // [B, L, D]
    const bf16* __restrict__ qprojT,  // [E, D]
    const bf16* __restrict__ kprojT,  // [E, D]
    bf16* __restrict__ qprime,        // [B, L, E]
    bf16* __restrict__ kprimeT,       // [B, E, L]
    float* __restrict__ qsum)         // [B, L]
{
  __shared__ float wsum[4][32];
  const int l0 = blockIdx.x * 32;
  const int b  = blockIdx.y;
  const bool isQ = (blockIdx.z == 0);
  const bf16* projT = isQ ? qprojT : kprojT;

  const int tid = threadIdx.x;
  const int wave = tid >> 5, lane = tid & 31;
  const int n = lane & 15, half = lane >> 4;

  const bf16* arow0 = xbf + ((size_t)b * LL + l0 + n) * DD + 16 * half;
  const bf16* arow1 = arow0 + (size_t)16 * DD;
  const bf16* bptr[4];
#pragma unroll
  for (int t = 0; t < 4; ++t)
    bptr[t] = projT + (size_t)(wave * 64 + t * 16 + n) * DD + 16 * half;

  v8f acc[2][4] = {};
  v16bf a0[2], a1[2], b0[4], b1[4];

#define LOAD_A(dst, k0) { dst[0] = load_afrag(arow0 + (k0)); dst[1] = load_afrag(arow1 + (k0)); }
#define LOAD_B(dst, k0) { _Pragma("unroll") for (int t = 0; t < 4; ++t) dst[t] = load_bfrag(bptr[t] + (k0)); }
#define STEP(aa, bb) { _Pragma("unroll") for (int t = 0; t < 4; ++t) { \
    acc[0][t] = wmma_bf16(aa[0], bb[t], acc[0][t]);                    \
    acc[1][t] = wmma_bf16(aa[1], bb[t], acc[1][t]); } }

  LOAD_A(a0, 0) LOAD_B(b0, 0)
  for (int k0 = 0; k0 < DD - 64; k0 += 64) {
    LOAD_A(a1, k0 + 32) LOAD_B(b1, k0 + 32)
    STEP(a0, b0)
    LOAD_A(a0, k0 + 64) LOAD_B(b0, k0 + 64)
    STEP(a1, b1)
  }
  LOAD_A(a1, DD - 32) LOAD_B(b1, DD - 32)
  STEP(a0, b0)
  STEP(a1, b1)

  // elementwise exp + per-row partial sums
  float rowpart[2][8];
#pragma unroll
  for (int rt = 0; rt < 2; ++rt)
#pragma unroll
    for (int r = 0; r < 8; ++r) rowpart[rt][r] = 0.0f;
#pragma unroll
  for (int rt = 0; rt < 2; ++rt)
#pragma unroll
    for (int t = 0; t < 4; ++t)
#pragma unroll
      for (int r = 0; r < 8; ++r) {
        const float v = __expf(acc[rt][t][r]);
        acc[rt][t][r] = v;
        rowpart[rt][r] += v;
      }

  if (isQ) {
    // q' row-major [L, E] (A-layout for stage 3)
#pragma unroll
    for (int rt = 0; rt < 2; ++rt)
#pragma unroll
      for (int t = 0; t < 4; ++t) {
        const int e0 = wave * 64 + t * 16;
#pragma unroll
        for (int r = 0; r < 8; ++r) {
          const int m = 16 * rt + 8 * half + r;
          qprime[((size_t)b * LL + l0 + m) * EE + e0 + n] = (bf16)acc[rt][t][r];
        }
      }
    // reduce over the 16 column-lanes (xor masks stay within each half)
#pragma unroll
    for (int rt = 0; rt < 2; ++rt)
#pragma unroll
      for (int r = 0; r < 8; ++r) {
        float s = rowpart[rt][r];
        s += __shfl_xor(s, 1);
        s += __shfl_xor(s, 2);
        s += __shfl_xor(s, 4);
        s += __shfl_xor(s, 8);
        rowpart[rt][r] = s;
      }
    if (n == 0) {
#pragma unroll
      for (int rt = 0; rt < 2; ++rt)
#pragma unroll
        for (int r = 0; r < 8; ++r)
          wsum[wave][16 * rt + 8 * half + r] = rowpart[rt][r];
    }
    __syncthreads();
    if (tid < 32) {
      const float s = wsum[0][tid] + wsum[1][tid] + wsum[2][tid] + wsum[3][tid];
      qsum[(size_t)b * LL + l0 + tid] = s;
    }
  } else {
    // k' transposed [E, L] (B-layout for stage 2): contiguous 16B per lane
#pragma unroll
    for (int rt = 0; rt < 2; ++rt)
#pragma unroll
      for (int t = 0; t < 4; ++t) {
        const int e0 = wave * 64 + t * 16;
        v8bf tv;
#pragma unroll
        for (int r = 0; r < 8; ++r) tv[r] = (bf16)acc[rt][t][r];
        bf16* dst = kprimeT + ((size_t)b * EE + e0 + n) * LL + l0 + 16 * rt + 8 * half;
        *(v8bf*)dst = tv;
      }
  }
#undef LOAD_A
#undef LOAD_B
#undef STEP
}

// ---- stage 2: kvT[b] = x^T @ k'  ([D,E], K = L = 4096) ---------------------
// block = 256 threads (8 waves); block tile = 64 (D) x 256 (E).
// LDS staged; global->reg prefetch of slab k+1 overlaps WMMAs of slab k.
__global__ __launch_bounds__(256) void kv_kernel(
    const bf16* __restrict__ xT,       // [B, D, L]  (A row-major [D x L])
    const bf16* __restrict__ kprimeT,  // [B, E, L]  (B^T row-major [E x L])
    bf16* __restrict__ kvT)            // [B, D, E]
{
  __shared__ bf16 aT[64][32];
  __shared__ bf16 bT[256][32];
  const int d0 = blockIdx.x * 64;
  const int b  = blockIdx.y;
  const int tid = threadIdx.x, wave = tid >> 5, lane = tid & 31;
  const int n = lane & 15, half = lane >> 4;
  const int dt = wave >> 1;              // d-tile 0..3 -> rows d0 + 16*dt
  const int eBase = (wave & 1) * 128;    // e range: 8 tiles of 16

  const bf16* aG = xT + ((size_t)b * DD + d0 + (tid >> 2)) * LL + (tid & 3) * 8;
  const bf16* bG = kprimeT + ((size_t)b * EE + tid) * LL;

  // prefetch first slab
  v8bf  ra = *(const v8bf*)(aG);
  v16bf rb = *(const v16bf*)(bG);

  v8f acc[8] = {};
  for (int k0 = 0; k0 < LL; k0 += 32) {
    __syncthreads();  // previous slab fully consumed
    *(v8bf*)(&aT[tid >> 2][(tid & 3) * 8]) = ra;
    *(v16bf*)(&bT[tid][0]) = rb;
    if (k0 + 32 < LL) {  // issue next-slab loads; latency hidden by WMMAs below
      ra = *(const v8bf*)(aG + k0 + 32);
      rb = *(const v16bf*)(bG + k0 + 32);
    }
    __syncthreads();

    const v16bf a = load_afrag(&aT[16 * dt + n][16 * half]);
#pragma unroll
    for (int t = 0; t < 8; ++t) {
      const v16bf bb = load_bfrag(&bT[eBase + 16 * t + n][16 * half]);
      acc[t] = wmma_bf16(a, bb, acc[t]);
    }
  }

#pragma unroll
  for (int t = 0; t < 8; ++t) {
    const int e0 = eBase + 16 * t;
#pragma unroll
    for (int r = 0; r < 8; ++r) {
      const int m = r + 8 * half;
      kvT[((size_t)b * DD + d0 + 16 * dt + m) * EE + e0 + n] = (bf16)acc[t][r];
    }
  }
}

// ---- stage 3: out = (q' @ kv) * 1/(rowsum(q')+1e-8)  (f32 out) -------------
// block = 128 threads (4 waves); tile = 32 (L) x 256 (D chunk); K = E = 256.
// Same 2-row-fragment, ping-pong structure as stage 1.
__global__ __launch_bounds__(128) void out_kernel(
    const bf16* __restrict__ qprime,  // [B, L, E]
    const bf16* __restrict__ kvT,     // [B, D, E]  (B^T row-major [D x E])
    const float* __restrict__ qsum,   // [B, L]
    float* __restrict__ out)          // [B, L, D]
{
  const int l0 = blockIdx.x * 32;
  const int dChunk = blockIdx.y * 256;
  const int b  = blockIdx.z;
  const int tid = threadIdx.x, wave = tid >> 5, lane = tid & 31;
  const int n = lane & 15, half = lane >> 4;

  const bf16* arow0 = qprime + ((size_t)b * LL + l0 + n) * EE + 16 * half;
  const bf16* arow1 = arow0 + (size_t)16 * EE;
  const bf16* bptr[4];
#pragma unroll
  for (int t = 0; t < 4; ++t)
    bptr[t] = kvT + ((size_t)b * DD + dChunk + wave * 64 + 16 * t + n) * EE + 16 * half;

  v8f acc[2][4] = {};
  v16bf a0[2], a1[2], b0[4], b1[4];

#define LOAD_A(dst, k0) { dst[0] = load_afrag(arow0 + (k0)); dst[1] = load_afrag(arow1 + (k0)); }
#define LOAD_B(dst, k0) { _Pragma("unroll") for (int t = 0; t < 4; ++t) dst[t] = load_bfrag(bptr[t] + (k0)); }
#define STEP(aa, bb) { _Pragma("unroll") for (int t = 0; t < 4; ++t) { \
    acc[0][t] = wmma_bf16(aa[0], bb[t], acc[0][t]);                    \
    acc[1][t] = wmma_bf16(aa[1], bb[t], acc[1][t]); } }

  LOAD_A(a0, 0) LOAD_B(b0, 0)
  for (int k0 = 0; k0 < EE - 64; k0 += 64) {
    LOAD_A(a1, k0 + 32) LOAD_B(b1, k0 + 32)
    STEP(a0, b0)
    LOAD_A(a0, k0 + 64) LOAD_B(b0, k0 + 64)
    STEP(a1, b1)
  }
  LOAD_A(a1, EE - 32) LOAD_B(b1, EE - 32)
  STEP(a0, b0)
  STEP(a1, b1)

#pragma unroll
  for (int rt = 0; rt < 2; ++rt)
#pragma unroll
    for (int r = 0; r < 8; ++r) {
      const int m = 16 * rt + 8 * half + r;
      const float z = 1.0f / (qsum[(size_t)b * LL + l0 + m] + 1e-8f);
      float* orow = out + ((size_t)b * LL + l0 + m) * DD;
#pragma unroll
      for (int t = 0; t < 4; ++t) {
        // streaming f32 output (134 MB): bypass caches
        __builtin_nontemporal_store(acc[rt][t][r] * z,
                                    &orow[dChunk + wave * 64 + 16 * t + n]);
      }
    }
#undef LOAD_A
#undef LOAD_B
#undef STEP
}

// ---------------------------------------------------------------------------
extern "C" void kernel_launch(void* const* d_in, const int* in_sizes, int n_in,
                              void* d_out, int out_size, void* d_ws, size_t ws_size,
                              hipStream_t stream) {
  const float* x  = (const float*)d_in[0];
  const float* qp = (const float*)d_in[1];
  const float* kp = (const float*)d_in[2];
  float* out = (float*)d_out;

  // workspace layout (bytes)
  char* ws = (char*)d_ws;
  const size_t szX   = (size_t)BB * LL * DD * 2;  // 64 MiB
  const size_t szP   = (size_t)EE * DD * 2;       // 512 KiB
  const size_t szPr  = (size_t)BB * LL * EE * 2;  // 16 MiB
  const size_t szKV  = (size_t)BB * DD * EE * 2;  // 4 MiB
  size_t off = 0;
  bf16* xbf     = (bf16*)(ws + off); off += szX;
  bf16* xT      = (bf16*)(ws + off); off += szX;
  bf16* qprojT  = (bf16*)(ws + off); off += szP;
  bf16* kprojT  = (bf16*)(ws + off); off += szP;
  bf16* qprime  = (bf16*)(ws + off); off += szPr;
  bf16* kprimeT = (bf16*)(ws + off); off += szPr;
  bf16* kvT     = (bf16*)(ws + off); off += szKV;
  float* qsum   = (float*)(ws + off); off += (size_t)BB * LL * 4;
  (void)ws_size; (void)in_sizes; (void)n_in; (void)out_size;

  prep_x_kernel   <<<dim3(DD / 32, LL / 32, BB), dim3(32, 8), 0, stream>>>(x, xbf, xT);
  prep_proj_kernel<<<dim3(EE / 32, DD / 32, 2),  dim3(32, 8), 0, stream>>>(qp, kp, qprojT, kprojT);
  proj_exp_kernel <<<dim3(LL / 32, BB, 2),       128, 0, stream>>>(xbf, qprojT, kprojT,
                                                                   qprime, kprimeT, qsum);
  kv_kernel       <<<dim3(DD / 64, BB),          256, 0, stream>>>(xT, kprimeT, kvT);
  out_kernel      <<<dim3(LL / 32, DD / 256, BB), 128, 0, stream>>>(qprime, kvT, qsum, out);
}
